// DiscreteLSTM_31215822307666
// MI455X (gfx1250) — matrix operand order repaired and checked by reading, not compiled, not run on hardware
//
#include <hip/hip_runtime.h>
#include <hip/hip_bf16.h>

typedef __attribute__((ext_vector_type(16))) __bf16 v16bf;
typedef __attribute__((ext_vector_type(8)))  __bf16 v8bf;
typedef __attribute__((ext_vector_type(8)))  float  v8f;

#define UNITS 1024
#define BATCH 128
#define SEQ   512

// ---------------------------------------------------------------------------
// Fragment loader: two contiguous 16-byte (8 x bf16) global loads -> v16bf.
// ---------------------------------------------------------------------------
__device__ __forceinline__ v16bf load_frag(const __bf16* p0, const __bf16* p1) {
    v8bf lo = *(const v8bf*)p0;
    v8bf hi = *(const v8bf*)p1;
    v16bf r;
#pragma unroll
    for (int j = 0; j < 8; ++j) { r[j] = lo[j]; r[8 + j] = hi[j]; }
    return r;
}

__device__ __forceinline__ float fast_sigmoid(float x) {
    return 1.0f / (1.0f + __expf(-x));
}
__device__ __forceinline__ float fast_tanh(float x) {
    float e = __expf(-2.0f * x);
    return (1.0f - e) / (1.0f + e);
}

// ---------------------------------------------------------------------------
// One-time: LDS-tiled transpose + convert 1024x1024 f32 W[k][n] -> bf16
// WT[n][k]. Coalesced on both the f32 read and the bf16 write.
// Block: 32x8 threads; each thread handles 4 rows of a 32x32 tile.
// ---------------------------------------------------------------------------
__global__ __launch_bounds__(256) void convert_transpose_kernel(
    const float* __restrict__ W, __bf16* __restrict__ WT)
{
    __shared__ float tile[32][33];               // +1 pad: no bank conflicts
    const int tx = threadIdx.x;                  // 0..31
    const int ty = threadIdx.y;                  // 0..7

    const int n_in = blockIdx.x * 32 + tx;       // contiguous in W rows
    const int k_in = blockIdx.y * 32 + ty;
#pragma unroll
    for (int j = 0; j < 4; ++j)
        tile[ty + 8 * j][tx] = W[(k_in + 8 * j) * UNITS + n_in];

    __syncthreads();

    const int k_out = blockIdx.y * 32 + tx;      // contiguous in WT rows
    const int n_out = blockIdx.x * 32 + ty;
#pragma unroll
    for (int j = 0; j < 4; ++j)
        WT[(n_out + 8 * j) * UNITS + k_out] = (__bf16)tile[tx][ty + 8 * j];
}

// One-time: zero the recurrent state (f32 + bf16 mirror).
__global__ __launch_bounds__(256) void init_state_kernel(
    float* __restrict__ h_f32, __bf16* __restrict__ h_bf)
{
    int idx = blockIdx.x * 256 + threadIdx.x;    // 0 .. BATCH*UNITS-1
    h_f32[idx] = 0.0f;
    h_bf[idx]  = (__bf16)0.0f;
}

// ---------------------------------------------------------------------------
// Step kernel 1:  inp = h @ Wh + E[x_t] + b   (written as bf16)
// grid = (UNITS/16, 2) blocks x 128 threads (4 waves): 512 waves spread over
// 128 WGPs; each wave owns one 16x16 tile. Two independent accumulator
// chains break the WMMA D->C dependency across the unrolled loop.
// ---------------------------------------------------------------------------
__global__ __launch_bounds__(128) void lstm_inp_kernel(
    const __bf16* __restrict__ h_bf, const __bf16* __restrict__ WhT,
    const float* __restrict__ E, const float* __restrict__ bias,
    const int* __restrict__ x, int t, __bf16* __restrict__ inp_bf)
{
    const int lane = threadIdx.x & 31;
    const int wave = threadIdx.x >> 5;           // 0..3
    const int n0 = blockIdx.x * 16;
    const int m0 = (blockIdx.y * 4 + wave) * 16;

    // A-fragment addressing (16x32 bf16, contiguous k = 16*(lane/16) + s)
    const int mA    = m0 + (lane & 15);
    const int koffA = (lane >> 4) * 16;
    // B-fragment addressing from transposed weights (lane owns column N)
    const int nB    = n0 + (lane & 15);
    const int koffB = (lane >> 4) * 16;

    const __bf16* arow = h_bf + mA * UNITS;
    const __bf16* brow = WhT + nB * UNITS;

    v8f acc0 = {}, acc1 = {};
#pragma unroll 2
    for (int k0 = 0; k0 < UNITS; k0 += 64) {
        v16bf a0 = load_frag(arow + k0 + koffA, arow + k0 + koffA + 8);
        v16bf b0 = load_frag(brow + k0 + koffB, brow + k0 + koffB + 8);
        acc0 = __builtin_amdgcn_wmma_f32_16x16x32_bf16(
                   false, a0, false, b0, (short)0, acc0, false, false);
        v16bf a1 = load_frag(arow + k0 + 32 + koffA, arow + k0 + 32 + koffA + 8);
        v16bf b1 = load_frag(brow + k0 + 32 + koffB, brow + k0 + 32 + koffB + 8);
        acc1 = __builtin_amdgcn_wmma_f32_16x16x32_bf16(
                   false, a1, false, b1, (short)0, acc1, false, false);
    }

    // Epilogue: C layout -> lane owns N = n0 + (lane&15), rows mBase + r
    const int nl    = n0 + (lane & 15);
    const int mBase = m0 + (lane >> 4) * 8;
    const float bn  = bias[nl];
#pragma unroll
    for (int r = 0; r < 8; ++r) {
        int m   = mBase + r;
        int tok = x[m * SEQ + t];
        float v = acc0[r] + acc1[r] + E[(long long)tok * UNITS + nl] + bn;
        inp_bf[m * UNITS + nl] = (__bf16)v;
    }
}

// ---------------------------------------------------------------------------
// Step kernel 2: all four gate GEMMs fused (shared A-fragment, 4 WMMAs per
// K-step on 4 independent accumulator chains), then the LSTM elementwise
// math; updates h (f32 + bf16 mirror) and writes y.
// ---------------------------------------------------------------------------
__global__ __launch_bounds__(128) void lstm_gate_kernel(
    const __bf16* __restrict__ inp_bf,
    const __bf16* __restrict__ WfT, const __bf16* __restrict__ WiT,
    const __bf16* __restrict__ WcT, const __bf16* __restrict__ WoT,
    const float* __restrict__ bfv, const float* __restrict__ biv,
    const float* __restrict__ bcv, const float* __restrict__ bov,
    float* __restrict__ h_f32, __bf16* __restrict__ h_bf,
    float* __restrict__ y, int t)
{
    const int lane = threadIdx.x & 31;
    const int wave = threadIdx.x >> 5;           // 0..3
    const int n0 = blockIdx.x * 16;
    const int m0 = (blockIdx.y * 4 + wave) * 16;

    const int mA    = m0 + (lane & 15);
    const int koffA = (lane >> 4) * 16;
    const int nB    = n0 + (lane & 15);
    const int koffB = (lane >> 4) * 16;

    const __bf16* arow = inp_bf + mA * UNITS;
    const __bf16* fr   = WfT + nB * UNITS;
    const __bf16* ir   = WiT + nB * UNITS;
    const __bf16* cr   = WcT + nB * UNITS;
    const __bf16* orow = WoT + nB * UNITS;

    v8f aF = {}, aI = {}, aC = {}, aO = {};
#pragma unroll 2
    for (int k0 = 0; k0 < UNITS; k0 += 32) {
        // prefetch next K panel of the shared A operand
        __builtin_prefetch(arow + k0 + 64, 0, 0);
        v16bf a = load_frag(arow + k0 + koffA, arow + k0 + koffA + 8);

        v16bf bF = load_frag(fr + k0 + koffB, fr + k0 + koffB + 8);
        aF = __builtin_amdgcn_wmma_f32_16x16x32_bf16(
                 false, a, false, bF, (short)0, aF, false, false);
        v16bf bI = load_frag(ir + k0 + koffB, ir + k0 + koffB + 8);
        aI = __builtin_amdgcn_wmma_f32_16x16x32_bf16(
                 false, a, false, bI, (short)0, aI, false, false);
        v16bf bC = load_frag(cr + k0 + koffB, cr + k0 + koffB + 8);
        aC = __builtin_amdgcn_wmma_f32_16x16x32_bf16(
                 false, a, false, bC, (short)0, aC, false, false);
        v16bf bO = load_frag(orow + k0 + koffB, orow + k0 + koffB + 8);
        aO = __builtin_amdgcn_wmma_f32_16x16x32_bf16(
                 false, a, false, bO, (short)0, aO, false, false);
    }

    const int nl    = n0 + (lane & 15);
    const int mBase = m0 + (lane >> 4) * 8;
    const float bfn = bfv[nl], bin = biv[nl], bcn = bcv[nl], bon = bov[nl];
#pragma unroll
    for (int r = 0; r < 8; ++r) {
        int m = mBase + r;
        float f = fast_sigmoid(aF[r] + bfn);
        float i = fast_sigmoid(aI[r] + bin);
        float c = fast_tanh   (aC[r] + bcn);
        float o = fast_sigmoid(aO[r] + bon);
        int hidx = m * UNITS + nl;
        float h1 = f * h_f32[hidx] + i * c;
        float yv = o * fast_tanh(h1);
        h_f32[hidx] = h1;
        h_bf[hidx]  = (__bf16)h1;
        y[(long long)m * (SEQ * UNITS) + (long long)t * UNITS + nl] = yv;
    }
}

// ---------------------------------------------------------------------------
// Host-side launch: convert weights once, zero state, then 512 (inp, gates)
// step pairs — all stream-ordered (graph-capture safe, deterministic).
// ---------------------------------------------------------------------------
extern "C" void kernel_launch(void* const* d_in, const int* in_sizes, int n_in,
                              void* d_out, int out_size, void* d_ws, size_t ws_size,
                              hipStream_t stream) {
    const int*   x   = (const int*)d_in[0];
    const float* E   = (const float*)d_in[1];
    const float* Wh  = (const float*)d_in[2];
    const float* b   = (const float*)d_in[3];
    const float* Wf  = (const float*)d_in[4];
    const float* bfv = (const float*)d_in[5];
    const float* Wi  = (const float*)d_in[6];
    const float* biv = (const float*)d_in[7];
    const float* Wc  = (const float*)d_in[8];
    const float* bcv = (const float*)d_in[9];
    const float* Wo  = (const float*)d_in[10];
    const float* bov = (const float*)d_in[11];
    float* out = (float*)d_out;

    // Workspace layout (≈ 11.0 MB total)
    const size_t WBYTES = (size_t)UNITS * UNITS * sizeof(__bf16);   // 2 MB
    const size_t SBF    = (size_t)BATCH * UNITS * sizeof(__bf16);   // 256 KB
    char* ws = (char*)d_ws;
    __bf16* WhT = (__bf16*)(ws + 0 * WBYTES);
    __bf16* WfT = (__bf16*)(ws + 1 * WBYTES);
    __bf16* WiT = (__bf16*)(ws + 2 * WBYTES);
    __bf16* WcT = (__bf16*)(ws + 3 * WBYTES);
    __bf16* WoT = (__bf16*)(ws + 4 * WBYTES);
    __bf16* inp_bf = (__bf16*)(ws + 5 * WBYTES);
    __bf16* h_bf   = (__bf16*)(ws + 5 * WBYTES + SBF);
    float*  h_f32  = (float*) (ws + 5 * WBYTES + 2 * SBF);

    dim3 convGrid(UNITS / 32, UNITS / 32);        // 32x32 tiles
    dim3 convBlock(32, 8);
    convert_transpose_kernel<<<convGrid, convBlock, 0, stream>>>(Wh, WhT);
    convert_transpose_kernel<<<convGrid, convBlock, 0, stream>>>(Wf, WfT);
    convert_transpose_kernel<<<convGrid, convBlock, 0, stream>>>(Wi, WiT);
    convert_transpose_kernel<<<convGrid, convBlock, 0, stream>>>(Wc, WcT);
    convert_transpose_kernel<<<convGrid, convBlock, 0, stream>>>(Wo, WoT);
    init_state_kernel<<<(BATCH * UNITS) / 256, 256, 0, stream>>>(h_f32, h_bf);

    dim3 stepGrid(UNITS / 16, 2);   // 64 N-tiles x 2 M-groups = 128 blocks
    for (int t = 0; t < SEQ; ++t) {
        lstm_inp_kernel<<<stepGrid, 128, 0, stream>>>(
            h_bf, WhT, E, b, x, t, inp_bf);
        lstm_gate_kernel<<<stepGrid, 128, 0, stream>>>(
            inp_bf, WfT, WiT, WcT, WoT, bfv, biv, bcv, bov,
            h_f32, h_bf, out, t);
    }
}
